// DimNetInteraction_48490180772447
// MI455X (gfx1250) — matrix operand order repaired and checked by reading, not compiled, not run on hardware
//
#include <hip/hip_runtime.h>

// ---------------------------------------------------------------------------
// DimeNet interaction block for gfx1250 (MI455X), wave32 + WMMA f16->f32,
// async global->LDS staging for B operands and the triplet gather.
// ---------------------------------------------------------------------------

typedef __attribute__((ext_vector_type(16))) _Float16 v16h;
typedef __attribute__((ext_vector_type(8)))  _Float16 v8h;
typedef __attribute__((ext_vector_type(8)))  float    v8f;

#define DIM  128
#define NBIL 8
#define NRAD 6
#define NSPH 7

__device__ __forceinline__ float silu_f(float v) {
    return v / (1.0f + __expf(-v));
}

// ---- CDNA5 async global->LDS copy (ASYNCcnt-tracked) ----------------------
__device__ __forceinline__ void async_b128(unsigned ldsoff, const void* gaddr) {
    asm volatile("global_load_async_to_lds_b128 %0, %1, off"
                 :: "v"(ldsoff), "v"(gaddr) : "memory");
}
__device__ __forceinline__ void wait_async0() {
    asm volatile("s_wait_asynccnt 0x0" ::: "memory");
}
__device__ __forceinline__ unsigned lds_off(const void* p) {
    return (unsigned)(uintptr_t)p;   // low 32 bits of a shared pointer = LDS offset
}

// ---------------------------------------------------------------------------
__device__ __forceinline__ v16h cvt16(float4 a0, float4 a1, float4 b0, float4 b1) {
    v16h v;
    v[0] = (_Float16)a0.x;  v[1] = (_Float16)a0.y;  v[2]  = (_Float16)a0.z;  v[3]  = (_Float16)a0.w;
    v[4] = (_Float16)a1.x;  v[5] = (_Float16)a1.y;  v[6]  = (_Float16)a1.z;  v[7]  = (_Float16)a1.w;
    v[8] = (_Float16)b0.x;  v[9] = (_Float16)b0.y;  v[10] = (_Float16)b0.z;  v[11] = (_Float16)b0.w;
    v[12] = (_Float16)b1.x; v[13] = (_Float16)b1.y; v[14] = (_Float16)b1.z;  v[15] = (_Float16)b1.w;
    return v;
}

// A fragment (16x32 f16, MxK): lane holds row m = lane%16; element order is
// K = kbase + g*8 + [0..7], then kbase + g*8 + 16 + [0..7]  (g = lane/16).
__device__ __forceinline__ v16h load_a_f32(const float* rowp, int koff) {
    float4 a0 = *(const float4*)(rowp + koff);
    float4 a1 = *(const float4*)(rowp + koff + 4);
    float4 b0 = *(const float4*)(rowp + koff + 16);
    float4 b1 = *(const float4*)(rowp + koff + 20);
    return cvt16(a0, a1, b0, b1);
}

// B fragment from LDS. Packed layout per fragment (512 halves = 1 KB):
//   chunk0: lane*16B at frag*1024 + lane*16        (elements 0..7,  K = k0+0..7)
//   chunk1: lane*16B at frag*1024 + 512 + lane*16  (elements 8..15, K = k0+16..23)
// Both DS reads are lane-contiguous 512 B sweeps -> bank-conflict-free.
__device__ __forceinline__ v16h load_b_frag_lds(const _Float16* sW, int frag, int lane) {
    const _Float16* p = sW + (size_t)frag * 512 + (size_t)lane * 8;
    v8h lo = *(const v8h*)(p);
    v8h hi = *(const v8h*)(p + 256);
    v16h r;
#pragma unroll
    for (int i = 0; i < 8; ++i) { r[i] = lo[i]; r[8 + i] = hi[i]; }
    return r;
}

__device__ __forceinline__ v8f wmma16(v16h a, v16h b, v8f c) {
    return __builtin_amdgcn_wmma_f32_16x16x32_f16(false, a, false, b, (short)0, c, false, false);
}

// ---------------------------------------------------------------------------
// Pack a K x N f32 weight matrix into the WMMA B-fragment f16 layout above.
// Element (k,n) read at W[k*rowStride + n*colStride].
// Fragment (kt,nt): lane l holds col n = nt*16 + l%16, K = kt*32 + (l/16)*8 + [0..7], then +16 + [0..7].
// ---------------------------------------------------------------------------
__global__ void pack_weights_k(const float* __restrict__ W, _Float16* __restrict__ dst,
                               int Ktot, int Ntot, int rowStride, int colStride) {
    int tid = blockIdx.x * blockDim.x + threadIdx.x;
    int KT = Ktot >> 5, NT = Ntot >> 4;
    int total = KT * NT * 32;
    if (tid >= total) return;
    int lane = tid & 31;
    int frag = tid >> 5;
    int nt = frag % NT;
    int kt = frag / NT;
    int n  = nt * 16 + (lane & 15);
    int k0 = kt * 32 + (lane >> 4) * 8;
    _Float16* o = dst + (size_t)frag * 512 + (size_t)lane * 8;
#pragma unroll
    for (int i = 0; i < 8; ++i) {
        o[i]       = (_Float16)W[(size_t)(k0 + i)     * rowStride + (size_t)n * colStride];
        o[256 + i] = (_Float16)W[(size_t)(k0 + 16 + i) * rowStride + (size_t)n * colStride];
    }
}

// rbf_p[e,c] = sum_r rbf[e,r] * W_rbf[r,c]
__global__ void rbf_proj_k(const float* __restrict__ rbf, const float* __restrict__ W,
                           float* __restrict__ out, int E) {
    int tid = blockIdx.x * blockDim.x + threadIdx.x;
    if (tid >= E * DIM) return;
    int e = tid >> 7, c = tid & 127;
    const float* r = rbf + (size_t)e * NRAD;
    float acc = 0.f;
#pragma unroll
    for (int i = 0; i < NRAD; ++i) acc += r[i] * W[i * DIM + c];
    out[tid] = acc;
}

// sbf_p[t,b] = sum_i sbf[t,i] * W_sbf[i,b]
__global__ void sbf_proj_k(const float* __restrict__ sbf, const float* __restrict__ W,
                           float* __restrict__ out, int T) {
    int tid = blockIdx.x * blockDim.x + threadIdx.x;
    if (tid >= T * NBIL) return;
    int t = tid >> 3, b = tid & 7;
    const float* s = sbf + (size_t)t * (NSPH * NRAD);
    float acc = 0.f;
#pragma unroll
    for (int i = 0; i < NSPH * NRAD; ++i) acc += s[i] * W[i * NBIL + b];
    out[tid] = acc;
}

// cooperative async copy of 32 KB packed weights into LDS (256 threads)
__device__ __forceinline__ void stage_w_256(_Float16* sW, const _Float16* gW, int tid) {
    unsigned    l = lds_off(sW) + tid * 128;
    const char* g = (const char*)gW + tid * 128;
#pragma unroll
    for (int i = 0; i < 8; ++i) { async_b128(l, g); l += 16; g += 16; }
}

// ---------------------------------------------------------------------------
// edge_embed: out_ji = silu(x@Wji + bji);  out_kj = silu(x@Wkj + bkj) * rbfp
// Dual-accumulator WMMA GEMM; both weight matrices staged in LDS per block.
// ---------------------------------------------------------------------------
__global__ __launch_bounds__(256) void edge_embed_k(
    const float* __restrict__ x,
    const _Float16* __restrict__ Wji, const _Float16* __restrict__ Wkj,
    const float* __restrict__ bji, const float* __restrict__ bkj,
    const float* __restrict__ rbfp,
    float* __restrict__ out_ji, float* __restrict__ out_kj, int nrows) {
    __shared__ _Float16 sWji[16384];
    __shared__ _Float16 sWkj[16384];
    int tid = threadIdx.x;
    stage_w_256(sWji, Wji, tid);
    stage_w_256(sWkj, Wkj, tid);
    wait_async0();
    __syncthreads();

    int lane = tid & 31;
    int wave = tid >> 5;
    int tile = blockIdx.x * 8 + wave;
    if (tile * 16 >= nrows) return;          // wave-uniform, after the barrier
    int row0 = tile * 16;
    int g    = lane >> 4;
    int mrow = lane & 15;
    const float* arow = x + (size_t)(row0 + mrow) * DIM;

    v8f cji[8] = {};
    v8f ckj[8] = {};
#pragma unroll
    for (int kt = 0; kt < 4; ++kt) {
        v16h a = load_a_f32(arow, kt * 32 + g * 8);
#pragma unroll
        for (int nt = 0; nt < 8; ++nt) {
            v16h b0 = load_b_frag_lds(sWji, kt * 8 + nt, lane);
            cji[nt] = wmma16(a, b0, cji[nt]);
            v16h b1 = load_b_frag_lds(sWkj, kt * 8 + nt, lane);
            ckj[nt] = wmma16(a, b1, ckj[nt]);
        }
    }
#pragma unroll
    for (int nt = 0; nt < 8; ++nt) {
        int col = nt * 16 + mrow;
        float bvj = bji[col];
        float bvk = bkj[col];
#pragma unroll
        for (int r = 0; r < 8; ++r) {
            int row = row0 + r + g * 8;
            size_t off = (size_t)row * DIM + col;
            out_ji[off] = silu_f(cji[nt][r] + bvj);
            out_kj[off] = silu_f(ckj[nt][r] + bvk) * rbfp[off];
        }
    }
}

// ---------------------------------------------------------------------------
// Generic WMMA GEMM:  out = silu(A@W + bias) [+ residual], W staged in LDS.
// ---------------------------------------------------------------------------
__global__ __launch_bounds__(256) void gemm_silu_k(
    const float* __restrict__ A, const _Float16* __restrict__ Wf,
    const float* __restrict__ bias, const float* __restrict__ residual,
    float* __restrict__ out, int nrows) {
    __shared__ _Float16 sW[16384];
    int tid = threadIdx.x;
    stage_w_256(sW, Wf, tid);
    wait_async0();
    __syncthreads();

    int lane = tid & 31;
    int wave = tid >> 5;
    int tile = blockIdx.x * 8 + wave;
    if (tile * 16 >= nrows) return;
    int row0 = tile * 16;
    int g    = lane >> 4;
    int mrow = lane & 15;
    const float* arow = A + (size_t)(row0 + mrow) * DIM;

    v8f acc[8] = {};
#pragma unroll
    for (int kt = 0; kt < 4; ++kt) {
        v16h a = load_a_f32(arow, kt * 32 + g * 8);
#pragma unroll
        for (int nt = 0; nt < 8; ++nt) {
            v16h b = load_b_frag_lds(sW, kt * 8 + nt, lane);
            acc[nt] = wmma16(a, b, acc[nt]);
        }
    }
#pragma unroll
    for (int nt = 0; nt < 8; ++nt) {
        int col = nt * 16 + mrow;
        float bv = bias[col];
#pragma unroll
        for (int r = 0; r < 8; ++r) {
            int row = row0 + r + g * 8;
            size_t off = (size_t)row * DIM + col;
            float v = silu_f(acc[nt][r] + bv);
            if (residual) v += residual[off];
            out[off] = v;
        }
    }
}

// ---------------------------------------------------------------------------
// Bilinear + segment-sum scatter.
//   m[t,:] = (sbfp[t,:] (x) t_kj[t,:]) @ W'   (K = 1024)
//   agg[edge_idx_ji[t], :] += m[t, :]         (global_atomic_add_f32)
// Block = 128 threads (4 waves), 64 triplets/block. Gather is async->LDS,
// W' streamed through LDS in 8 chunks of 32 KB (4 k-steps each).
// ---------------------------------------------------------------------------
__global__ __launch_bounds__(128) void bilinear_scatter_k(
    const float* __restrict__ xkj, const float* __restrict__ sbfp,
    const int* __restrict__ idx_kj, const int* __restrict__ idx_ji,
    const _Float16* __restrict__ Wbf, float* __restrict__ agg, int T) {
    __shared__ _Float16 sW[16384];      // 32 KB chunk of packed W'
    __shared__ float ld_t[64 * DIM];    // gathered t_kj tile (f32)
    __shared__ float ld_s[64 * NBIL];   // sbf_p tile
    __shared__ int   ld_ji[64];

    int t0  = blockIdx.x * 64;
    int tid = threadIdx.x;

    // ---- gather stage (async global->LDS), 64 floats per thread ---------
    {
        int row  = tid >> 1;   // 0..63
        int half = tid & 1;    // 0/1
        int trip = t0 + row;
        if (trip >= T) trip = T - 1;   // tail rows: real data, scatter is masked
        int e = idx_kj[trip];
        unsigned    l = lds_off(ld_t + row * DIM + half * 64);
        const char* g = (const char*)(xkj + (size_t)e * DIM + half * 64);
#pragma unroll
        for (int i = 0; i < 16; ++i) { async_b128(l, g); l += 16; g += 16; }
        async_b128(lds_off(ld_s + row * NBIL + half * 4),
                   sbfp + (size_t)trip * NBIL + half * 4);
        if (half == 0) ld_ji[row] = idx_ji[trip];
    }
    wait_async0();
    __syncthreads();

    int lane = tid & 31;
    int wave = tid >> 5;       // 0..3, each wave owns 16 triplet rows
    int g    = lane >> 4;
    int mrow = lane & 15;
    int rloc = wave * 16 + mrow;
    const float* trow = ld_t + rloc * DIM;
    const float* srow = ld_s + rloc * NBIL;

    v8f acc[8] = {};
    for (int c = 0; c < 8; ++c) {           // 8 chunks x 4 k-steps = K 1024
        // stage W' chunk c: 32 KB / 128 threads = 256 B per thread
        {
            unsigned    l = lds_off(sW) + tid * 256;
            const char* gp = (const char*)Wbf + c * 32768 + tid * 256;
#pragma unroll
            for (int i = 0; i < 16; ++i) { async_b128(l, gp); l += 16; gp += 16; }
        }
        wait_async0();
        __syncthreads();
#pragma unroll
        for (int kl = 0; kl < 4; ++kl) {
            int   kt = c * 4 + kl;
            int   b  = kt >> 2;             // sbf channel
            float s  = srow[b];
            int   j0 = (kt & 3) * 32 + g * 8;
            float4 a0 = *(const float4*)(trow + j0);
            float4 a1 = *(const float4*)(trow + j0 + 4);
            float4 b0 = *(const float4*)(trow + j0 + 16);
            float4 b1 = *(const float4*)(trow + j0 + 20);
            v16h a;
            a[0]  = (_Float16)(s * a0.x); a[1]  = (_Float16)(s * a0.y);
            a[2]  = (_Float16)(s * a0.z); a[3]  = (_Float16)(s * a0.w);
            a[4]  = (_Float16)(s * a1.x); a[5]  = (_Float16)(s * a1.y);
            a[6]  = (_Float16)(s * a1.z); a[7]  = (_Float16)(s * a1.w);
            a[8]  = (_Float16)(s * b0.x); a[9]  = (_Float16)(s * b0.y);
            a[10] = (_Float16)(s * b0.z); a[11] = (_Float16)(s * b0.w);
            a[12] = (_Float16)(s * b1.x); a[13] = (_Float16)(s * b1.y);
            a[14] = (_Float16)(s * b1.z); a[15] = (_Float16)(s * b1.w);
#pragma unroll
            for (int nt = 0; nt < 8; ++nt) {
                v16h bb = load_b_frag_lds(sW, kl * 8 + nt, lane);
                acc[nt] = wmma16(a, bb, acc[nt]);
            }
        }
        __syncthreads();    // all waves done reading before chunk overwrite
    }

    // ---- scatter-add (segment sum) via float atomics ---------------------
#pragma unroll
    for (int nt = 0; nt < 8; ++nt) {
        int col = nt * 16 + mrow;
#pragma unroll
        for (int r = 0; r < 8; ++r) {
            int lrow = wave * 16 + r + g * 8;
            if (t0 + lrow < T) {
                int e = ld_ji[lrow];
                atomicAdd(agg + (size_t)e * DIM + col, acc[nt][r]);
            }
        }
    }
}

// ---------------------------------------------------------------------------
// Launcher
// ---------------------------------------------------------------------------
extern "C" void kernel_launch(void* const* d_in, const int* in_sizes, int n_in,
                              void* d_out, int out_size, void* d_ws, size_t ws_size,
                              hipStream_t stream) {
    (void)in_sizes; (void)n_in; (void)out_size; (void)ws_size;
    const float* x     = (const float*)d_in[0];
    const float* rbf   = (const float*)d_in[1];
    const float* sbf   = (const float*)d_in[2];
    const int*   ikj   = (const int*)d_in[3];
    const int*   iji   = (const int*)d_in[4];
    const float* W_rbf = (const float*)d_in[5];
    const float* W_sbf = (const float*)d_in[6];
    const float* W_kj  = (const float*)d_in[7];
    const float* b_kj  = (const float*)d_in[8];
    const float* W_ji  = (const float*)d_in[9];
    const float* b_ji  = (const float*)d_in[10];
    const float* W_bil = (const float*)d_in[11];
    const float* W_res = (const float*)d_in[12];
    const float* b_res = (const float*)d_in[13];
    const float* W_out = (const float*)d_in[14];
    const float* b_out = (const float*)d_in[15];

    const int E = 150000;
    const int T = 450000;

    // ---- workspace carve-up --------------------------------------------
    char* ws = (char*)d_ws;
    _Float16* fr    = (_Float16*)ws;
    _Float16* fWji  = fr;                       // 16384 halves each 128x128
    _Float16* fWkj  = fr + 16384;
    _Float16* fWout = fr + 32768;
    _Float16* fWres = fr + 49152;               // 6 x 16384
    _Float16* fWbil = fr + 49152 + 6 * 16384;   // 1024x128 -> 131072 halves
    size_t fbytes = (size_t)(49152 + 6 * 16384 + 131072) * sizeof(_Float16); // 557056

    float* bufA = (float*)(ws + fbytes);        // rbf_p, later t-buffer
    float* bufB = bufA + (size_t)E * DIM;       // x_kj, later h-buffer
    float* sbfp = bufB + (size_t)E * DIM;       // (T, 8)
    float* AGG  = (float*)d_out;                // x_ji + segment_sum, later h3, final out

    // ---- weight packing -------------------------------------------------
    pack_weights_k<<<4, 256, 0, stream>>>(W_ji,  fWji,  128, 128, 128, 1);
    pack_weights_k<<<4, 256, 0, stream>>>(W_kj,  fWkj,  128, 128, 128, 1);
    pack_weights_k<<<4, 256, 0, stream>>>(W_out, fWout, 128, 128, 128, 1);
    for (int k = 0; k < 6; ++k)
        pack_weights_k<<<4, 256, 0, stream>>>(W_res + (size_t)k * 16384,
                                              fWres + (size_t)k * 16384,
                                              128, 128, 128, 1);
    // W'[(b,j), o] = W_bil[o*1024 + b*128 + j]  ->  rowStride=1, colStride=1024
    pack_weights_k<<<32, 256, 0, stream>>>(W_bil, fWbil, 1024, 128, 1, 1024);

    // ---- small projections ---------------------------------------------
    rbf_proj_k<<<(E * DIM + 255) / 256, 256, 0, stream>>>(rbf, W_rbf, bufA, E);
    sbf_proj_k<<<(T * NBIL + 255) / 256, 256, 0, stream>>>(sbf, W_sbf, sbfp, T);

    int blocksE = ((E / 16) + 7) / 8;   // 1172 blocks of 8 waves

    // ---- edge embedding: AGG = x_ji, bufB = x_kj ------------------------
    edge_embed_k<<<blocksE, 256, 0, stream>>>(x, fWji, fWkj, b_ji, b_kj,
                                              bufA, AGG, bufB, E);

    // ---- bilinear + segment-sum into AGG --------------------------------
    bilinear_scatter_k<<<(T + 63) / 64, 128, 0, stream>>>(bufB, sbfp, ikj, iji,
                                                          fWbil, AGG, T);

    // ---- residual chain -------------------------------------------------
    // t1 = silu(AGG @ Wres00 + b00)               -> bufA
    gemm_silu_k<<<blocksE, 256, 0, stream>>>(AGG,  fWres,             b_res,       nullptr, bufA, E);
    // h2 = AGG + silu(t1 @ Wres01 + b01)          -> bufB
    gemm_silu_k<<<blocksE, 256, 0, stream>>>(bufA, fWres + 16384,     b_res + 128, AGG,     bufB, E);
    // h3 = x + silu(h2 @ W_out + b_out)           -> AGG (d_out)
    gemm_silu_k<<<blocksE, 256, 0, stream>>>(bufB, fWout,             b_out,       x,       AGG,  E);
    // t2 = silu(h3 @ Wres10 + b10)                -> bufA
    gemm_silu_k<<<blocksE, 256, 0, stream>>>(AGG,  fWres + 2 * 16384, b_res + 256, nullptr, bufA, E);
    // h4 = h3 + silu(t2 @ Wres11 + b11)           -> bufB
    gemm_silu_k<<<blocksE, 256, 0, stream>>>(bufA, fWres + 3 * 16384, b_res + 384, AGG,     bufB, E);
    // t3 = silu(h4 @ Wres20 + b20)                -> bufA
    gemm_silu_k<<<blocksE, 256, 0, stream>>>(bufB, fWres + 4 * 16384, b_res + 512, nullptr, bufA, E);
    // out = h4 + silu(t3 @ Wres21 + b21)          -> d_out
    gemm_silu_k<<<blocksE, 256, 0, stream>>>(bufA, fWres + 5 * 16384, b_res + 640, bufB,    AGG,  E);
}
